// SubstitutionEmbedding_45956150067988
// MI455X (gfx1250) — matrix operand order repaired and checked by reading, not compiled, not run on hardware
//
#include <hip/hip_runtime.h>
#include <hip/hip_bf16.h>
#include <limits.h>
#include <stdint.h>

// ---------------- problem constants (from reference) ----------------
constexpr int kB    = 4;
constexpr int kL1   = 65536;        // padded penultimate length
constexpr int kL2   = 262144;       // padded last-layer length
constexpr int kSEQ  = kL1 + kL2;    // 327680
constexpr int kG2   = kL2 / 8;      // 32768 conv2 output rows per batch
constexpr int kM1   = kL1 / 8;      // 8192  conv1 output rows per batch
constexpr int kCD   = 32;           // conv depth
constexpr int kED   = 256;          // embed dim
constexpr int kK    = 256;          // GEMM reduction dim (= kCD * 8)

// ---------------- WMMA / TDM types ----------------
typedef __attribute__((ext_vector_type(16))) __bf16       v16bf;
typedef __attribute__((ext_vector_type(8)))  float        v8f;
typedef __attribute__((ext_vector_type(4)))  unsigned int u32x4;
typedef __attribute__((ext_vector_type(4)))  unsigned int uint32x4;
typedef __attribute__((ext_vector_type(4)))  int          int32x4;
typedef __attribute__((ext_vector_type(8)))  int          int32x8;

union V16 {
    v16bf          v;
    unsigned short u[16];
    u32x4          q[2];
};

__device__ inline unsigned short bf16_rne(float f) {
    union { float f; unsigned int u; } x;
    x.f = f;
    unsigned int u = x.u;
    u += 0x7FFFu + ((u >> 16) & 1u);   // round-to-nearest-even
    return (unsigned short)(u >> 16);
}

// ---------------- Tensor Data Mover (gfx1250) ----------------
#ifndef __has_builtin
#define __has_builtin(x) 0
#endif
#if __has_builtin(__builtin_amdgcn_tensor_load_to_lds) && \
    __has_builtin(__builtin_amdgcn_s_wait_tensorcnt)
#define USE_TDM 1
#else
#define USE_TDM 0
#endif

#if USE_TDM
// Flat LDS addresses carry the aperture in bits[63:32]; bits[31:0] are the
// LDS byte offset (ISA 10.2 aperture table), so truncation yields D#.lds_addr.
__device__ inline unsigned lds_off_u32(const void* p) {
    return (unsigned)(unsigned long long)(uintptr_t)p;
}

// 2-D tile load, 2-byte elements: LDS[y][x] (row-major, contiguous) =
// MEM[gaddr + 2*(x + y*stride0)], x < tile_d0, y < tile_d1.
// D# layout per CDNA5 ISA; groups 2-3 zero (<=2D).
__device__ inline void tdm_load_2d_b16(unsigned lds_byte, unsigned long long gaddr,
                                       unsigned tile_d0, unsigned tile_d1,
                                       unsigned long long stride0,
                                       unsigned tensor_d0, unsigned tensor_d1) {
    uint32x4 g0;
    g0.x = 1u;                                                // count=1 (valid), user mode
    g0.y = lds_byte;                                          // lds_addr [63:32]
    g0.z = (unsigned)(gaddr & 0xffffffffull);                 // global_addr[31:0]
    g0.w = (unsigned)((gaddr >> 32) & 0x01ffffffull) | (2u << 30);  // [56:32] | type=2
    int32x8 g1 = {};
    g1[0] = (int)(1u << 16);                                  // wg_mask=0, data_size=1 (2B)
    g1[1] = (int)((tensor_d0 & 0xffffu) << 16);               // tensor_dim0[15:0] @ [63:48]
    g1[2] = (int)(((tensor_d0 >> 16) & 0xffffu) | ((tensor_d1 & 0xffffu) << 16));
    g1[3] = (int)(((tensor_d1 >> 16) & 0xffffu) | ((tile_d0 & 0xffffu) << 16));
    g1[4] = (int)(tile_d1 & 0xffffu);                         // tile_dim1; tile_dim2=0
    g1[5] = (int)(unsigned)(stride0 & 0xffffffffull);         // tensor_dim0_stride[31:0]
    g1[6] = (int)(unsigned)((stride0 >> 32) & 0xffffull);     // [47:32]; dim1_stride=0
    g1[7] = 0;
    int32x4 gz = {};
#if __clang_major__ >= 23
    int32x8 gz8 = {};
    __builtin_amdgcn_tensor_load_to_lds(g0, g1, gz, gz, gz8, 0);
#else
    __builtin_amdgcn_tensor_load_to_lds(g0, g1, gz, gz, 0);
#endif
}
#endif  // USE_TDM

// ---------------- stage 0: max depth + layer lengths ----------------
__global__ void k_init(int* maxd, int* len1, int* len2) {
    if (threadIdx.x == 0) *maxd = INT_MIN;
    if (threadIdx.x < kB) { len1[threadIdx.x] = 0; len2[threadIdx.x] = 0; }
}

__global__ void k_maxdepth(const int* __restrict__ depth, int n, int* __restrict__ maxd) {
    __shared__ int s[256];
    int lm = INT_MIN;
    for (int i = blockIdx.x * blockDim.x + threadIdx.x; i < n; i += gridDim.x * blockDim.x)
        lm = max(lm, depth[i]);
    s[threadIdx.x] = lm;
    __syncthreads();
    for (int o = 128; o > 0; o >>= 1) {
        if (threadIdx.x < o) s[threadIdx.x] = max(s[threadIdx.x], s[threadIdx.x + o]);
        __syncthreads();
    }
    if (threadIdx.x == 0) atomicMax(maxd, s[0]);
}

__global__ void k_lens(const int* __restrict__ depth, const int* __restrict__ maxd,
                       int* __restrict__ len1, int* __restrict__ len2) {
    __shared__ int c1[kB], c2[kB];
    if (threadIdx.x < kB) { c1[threadIdx.x] = 0; c2[threadIdx.x] = 0; }
    __syncthreads();
    const int md = *maxd;
    const int n  = kB * kSEQ;
    for (int i = blockIdx.x * blockDim.x + threadIdx.x; i < n; i += gridDim.x * blockDim.x) {
        int b = i / kSEQ;
        int d = depth[i];
        if (d == md - 1)      atomicAdd(&c1[b], 1);
        else if (d == md)     atomicAdd(&c2[b], 1);
    }
    __syncthreads();
    if (threadIdx.x < kB) {
        if (c1[threadIdx.x]) atomicAdd(&len1[threadIdx.x], c1[threadIdx.x]);
        if (c2[threadIdx.x]) atomicAdd(&len2[threadIdx.x], c2[threadIdx.x]);
    }
}

// ---------------- stage 1: substitution index computation ----------------
__global__ __launch_bounds__(1024)
void k_scan(const int* __restrict__ value, const int* __restrict__ len1a,
            const int* __restrict__ len2a, int* __restrict__ grank,
            int* __restrict__ srcidx, unsigned char* __restrict__ tok1) {
    __shared__ int sd[1024];
    const int b    = blockIdx.x;
    const int tid  = threadIdx.x;
    const int len1 = len1a[b];
    const int len2 = len2a[b];
    const int* vrow = value + (size_t)b * kSEQ;
    int*       gr   = grank + (size_t)b * kG2;

    for (int i = tid; i < kG2; i += 1024) gr[i] = kG2 - 1;   // padded order entries, pre-clipped
    __syncthreads();

    // ---- phase A: group mask scan + scatter ----
    constexpr int CA = kG2 / 1024;   // 32
    int cnt = 0;
    for (int c = 0; c < CA; ++c) {
        int g = tid * CA + c;
        int j = g * 8;
        int v = 0;
        if (j < len2) { int gi = len1 + j; if (gi > kSEQ - 1) gi = kSEQ - 1; v = vrow[gi]; }
        cnt += (v != 0);
    }
    sd[tid] = cnt;
    __syncthreads();
    for (int o = 1; o < 1024; o <<= 1) {
        int t = (tid >= o) ? sd[tid - o] : 0;
        __syncthreads();
        sd[tid] += t;
        __syncthreads();
    }
    int run = sd[tid] - cnt;   // exclusive prefix
    for (int c = 0; c < CA; ++c) {
        int g = tid * CA + c;
        int j = g * 8;
        int v = 0;
        if (j < len2) { int gi = len1 + j; if (gi > kSEQ - 1) gi = kSEQ - 1; v = vrow[gi]; }
        if (v != 0) { gr[run] = g; ++run; }
    }
    __syncthreads();

    // ---- phase B: mixed-token scan -> src ----
    constexpr int CB = kL1 / 1024;   // 64
    cnt = 0;
    for (int c = 0; c < CB; ++c) {
        int j = tid * CB + c;
        int v = (j < len1) ? vrow[j] : 0;
        cnt += (v == 2);
    }
    sd[tid] = cnt;
    __syncthreads();
    for (int o = 1; o < 1024; o <<= 1) {
        int t = (tid >= o) ? sd[tid - o] : 0;
        __syncthreads();
        sd[tid] += t;
        __syncthreads();
    }
    run = sd[tid] - cnt;
    for (int c = 0; c < CB; ++c) {
        int j = tid * CB + c;
        int v = (j < len1) ? vrow[j] : 0;
        if (v < 0) v = 0; if (v > 3) v = 3;
        tok1[(size_t)b * kL1 + j] = (unsigned char)v;
        int ismix = (v == 2);
        int rank  = run + ismix - 1;
        if (rank < 0) rank = 0;
        if (rank > kG2 - 1) rank = kG2 - 1;
        srcidx[(size_t)b * kL1 + j] = gr[rank];
        run += ismix;
    }
}

__global__ void k_tok2(const int* __restrict__ value, const int* __restrict__ len1a,
                       const int* __restrict__ len2a, unsigned char* __restrict__ tok2) {
    const long long n = (long long)kB * kL2;
    for (long long i = blockIdx.x * (long long)blockDim.x + threadIdx.x; i < n;
         i += (long long)gridDim.x * blockDim.x) {
        int b = (int)(i / kL2), j = (int)(i % kL2);
        int v = 0;
        if (j < len2a[b]) {
            int gi = len1a[b] + j;
            if (gi > kSEQ - 1) gi = kSEQ - 1;
            v = value[(size_t)b * kSEQ + gi];
        }
        if (v < 0) v = 0; if (v > 3) v = 3;
        tok2[i] = (unsigned char)v;
    }
}

// ---------------- stage 2: f32 -> bf16 weight/embedding tables ----------------
__global__ void k_wconv(const float* __restrict__ w1, const float* __restrict__ w2,
                        const float* __restrict__ e1, const float* __restrict__ e2,
                        unsigned short* __restrict__ W1bf, unsigned short* __restrict__ W2bf,
                        unsigned short* __restrict__ E1bf, unsigned short* __restrict__ E2bf) {
    const int n = kED * kK + kCD * kK + 128 + 128;
    for (int i = blockIdx.x * blockDim.x + threadIdx.x; i < n; i += gridDim.x * blockDim.x) {
        if (i < kED * kK)                       W1bf[i]              = bf16_rne(w1[i]);
        else if (i < kED * kK + kCD * kK)       W2bf[i - kED * kK]   = bf16_rne(w2[i - kED * kK]);
        else if (i < kED * kK + kCD * kK + 128) E1bf[i - kED*kK - kCD*kK]       = bf16_rne(e1[i - kED*kK - kCD*kK]);
        else                                    E2bf[i - kED*kK - kCD*kK - 128] = bf16_rne(e2[i - kED*kK - kCD*kK - 128]);
    }
}

// ---------------- stage 3: conv2 GEMM (M=kG2/batch, K=256, N=32), bf16 WMMA ----------------
// Block: 256 thr = 8 waves; 64 output rows; waves = 4 row-tiles x 2 col-tiles.
// W2 panel (16 KB, contiguous) fetched by the Tensor Data Mover while the wave
// builds the A panel from tokens; wave 0 waits TENSORcnt, barrier publishes.
__global__ __launch_bounds__(256)
void k_gemm2(const unsigned char* __restrict__ tok2, const unsigned short* __restrict__ W2bf,
             const unsigned short* __restrict__ E2bf, const float* __restrict__ bias2,
             float* __restrict__ y) {
    __shared__ __align__(16) unsigned short sA[64 * kK];   // A panel, fragment order [row][p]
    __shared__ __align__(16) unsigned short sW[kCD * kK];  // W2 bf16, layout [co][p]
    __shared__ unsigned char sT[64 * 8];
    __shared__ unsigned short sE[128];

    const int b   = blockIdx.y;
    const int g0  = blockIdx.x * 64;
    const int tid = threadIdx.x;

#if USE_TDM
    if (tid < 32) {
        tdm_load_2d_b16(lds_off_u32(&sW[0]),
                        (unsigned long long)(uintptr_t)W2bf,
                        /*tile_d0=*/kCD * kK, /*tile_d1=*/1,
                        /*stride0=*/kCD * kK,
                        /*tensor_d0=*/kCD * kK, /*tensor_d1=*/1);
    }
#else
    for (int i = tid; i < kCD * kK; i += 256) sW[i] = W2bf[i];
#endif
    for (int i = tid; i < 128; i += 256) sE[i] = E2bf[i];
    {
        const unsigned char* tp = tok2 + (size_t)b * kL2 + (size_t)g0 * 8;
        for (int i = tid; i < 512; i += 256) sT[i] = tp[i];
    }
    __syncthreads();
    // Build A panel: sA[rb*256 + ci*8 + kk] = emb2[tok[rb,kk]][ci]
    for (int i = tid; i < 64 * kK; i += 256) {
        int rb = i >> 8, p = i & 255;
        int kk = p & 7, ci = p >> 3;
        sA[i] = sE[sT[rb * 8 + kk] * 32 + ci];
    }
#if USE_TDM
    if (tid < 32) __builtin_amdgcn_s_wait_tensorcnt((short)0);
#endif
    __syncthreads();

    const int w    = tid >> 5, lane = tid & 31;
    const int wr   = w >> 1,   wc   = w & 1;
    const int m    = lane & 15;
    const int hi   = lane >> 4;     // 0|1
    const int kh   = hi * 8;        // A K-half offset
    v8f acc = {};
    #pragma unroll
    for (int t = 0; t < 8; ++t) {
        V16 a, bf;
        int abase = (wr * 16 + m) * kK + t * 32 + kh;
        a.q[0]  = *reinterpret_cast<const u32x4*>(&sA[abase]);
        a.q[1]  = *reinterpret_cast<const u32x4*>(&sA[abase + 16]);
        int bbase = (wc * 16 + m) * kK + t * 32 + hi * 8;
        bf.q[0] = *reinterpret_cast<const u32x4*>(&sW[bbase]);
        bf.q[1] = *reinterpret_cast<const u32x4*>(&sW[bbase + 16]);
        acc = __builtin_amdgcn_wmma_f32_16x16x32_bf16(false, a.v, false, bf.v,
                                                      (short)0, acc, false, false);
    }
    #pragma unroll
    for (int r = 0; r < 8; ++r) {
        int row = g0 + wr * 16 + r + hi * 8;
        int col = wc * 16 + m;
        y[((size_t)b * kG2 + row) * kCD + col] = acc[r] + bias2[col];
    }
}

// ---------------- stage 4: substitution + conv1 GEMM (M=kM1/batch, K=256, N=256) ----------------
// Block: 256 thr = 8 waves; 64 output rows x 256 cols; waves = 4 row-tiles x 2 col-halves.
// W1 k-slabs (2-D tiles: 256 rows x 64 B, 512 B stride) are double-buffered in LDS and
// prefetched with the TDM one step ahead (issue slab t+1, s_wait_tensorcnt 1 -> slab t done).
__global__ __launch_bounds__(256)
void k_gemm1(const unsigned char* __restrict__ tok1, const int* __restrict__ srcidx,
             const unsigned short* __restrict__ W1bf, const unsigned short* __restrict__ E1bf,
             const float* __restrict__ y, const float* __restrict__ bias1,
             float* __restrict__ out) {
    __shared__ __align__(16) unsigned short sA[64 * kK];      // A panel [row][p], 32 KB
    __shared__ __align__(16) unsigned short sW[2][kED * 32];  // W1 k-slab x2, [co][kk], 2x16 KB

    const int b   = blockIdx.y;
    const int g0  = blockIdx.x * 64;
    const int tid = threadIdx.x;

#if USE_TDM
    if (tid < 32) {
        tdm_load_2d_b16(lds_off_u32(&sW[0][0]),
                        (unsigned long long)(uintptr_t)W1bf,
                        /*tile_d0=*/32, /*tile_d1=*/kED, /*stride0=*/kK,
                        /*tensor_d0=*/kK, /*tensor_d1=*/kED);
    }
#endif

    // Stage x rows (512 of them) directly in A-fragment order:
    // sA[(jr>>3)*256 + ci*8 + (jr&7)] = x[jr][ci]
    for (int rep = 0; rep < 2; ++rep) {
        int jr = tid + rep * 256;
        int j  = g0 * 8 + jr;
        int gl = jr >> 3, kk = jr & 7;
        unsigned short* dst = &sA[gl * kK + kk];
        int tok = tok1[(size_t)b * kL1 + j];
        if (tok == 2) {
            const float* yp = y + ((size_t)b * kG2 + srcidx[(size_t)b * kL1 + j]) * kCD;
            for (int ci = 0; ci < kCD; ++ci) dst[ci * 8] = bf16_rne(yp[ci]);
        } else {
            for (int ci = 0; ci < kCD; ++ci) dst[ci * 8] = E1bf[tok * 32 + ci];
        }
    }

    const int w  = tid >> 5, lane = tid & 31;
    const int wr = w >> 1,   wc   = w & 1;
    const int m  = lane & 15;
    const int hi = lane >> 4;
    const int kh = hi * 8;
    v8f acc[8];
    #pragma unroll
    for (int i = 0; i < 8; ++i) acc[i] = (v8f){};

    for (int t = 0; t < 8; ++t) {
#if USE_TDM
        if (tid < 32) {
            if (t < 7) {
                tdm_load_2d_b16(lds_off_u32(&sW[(t + 1) & 1][0]),
                                (unsigned long long)(uintptr_t)(W1bf + (size_t)(t + 1) * 32),
                                /*tile_d0=*/32, /*tile_d1=*/kED, /*stride0=*/kK,
                                /*tensor_d0=*/kK, /*tensor_d1=*/kED);
                __builtin_amdgcn_s_wait_tensorcnt((short)1);   // slab t complete
            } else {
                __builtin_amdgcn_s_wait_tensorcnt((short)0);
            }
        }
        __syncthreads();   // publish slab t (and A panel on t==0)
#else
        for (int i = tid; i < kED * 32; i += 256) {
            int co = i >> 5, kk = i & 31;
            sW[t & 1][i] = W1bf[co * kK + t * 32 + kk];
        }
        __syncthreads();
#endif
        const unsigned short* sWc = sW[t & 1];

        V16 a;
        int abase = (wr * 16 + m) * kK + t * 32 + kh;
        a.q[0] = *reinterpret_cast<const u32x4*>(&sA[abase]);
        a.q[1] = *reinterpret_cast<const u32x4*>(&sA[abase + 16]);
        #pragma unroll
        for (int nt = 0; nt < 8; ++nt) {
            V16 bf;
            int col   = wc * 128 + nt * 16 + m;
            int bbase = col * 32 + hi * 8;
            bf.q[0] = *reinterpret_cast<const u32x4*>(&sWc[bbase]);
            bf.q[1] = *reinterpret_cast<const u32x4*>(&sWc[bbase + 16]);
            acc[nt] = __builtin_amdgcn_wmma_f32_16x16x32_bf16(false, a.v, false, bf.v,
                                                              (short)0, acc[nt], false, false);
        }
        __syncthreads();   // all reads of slab t done before its buffer is overwritten
    }

    #pragma unroll
    for (int nt = 0; nt < 8; ++nt) {
        #pragma unroll
        for (int r = 0; r < 8; ++r) {
            int row = g0 + wr * 16 + r + hi * 8;
            int col = wc * 128 + nt * 16 + m;
            out[((size_t)b * kM1 + row) * kED + col] = acc[nt][r] + bias1[col];
        }
    }
}

// ---------------- host launcher ----------------
extern "C" void kernel_launch(void* const* d_in, const int* in_sizes, int n_in,
                              void* d_out, int out_size, void* d_ws, size_t ws_size,
                              hipStream_t stream) {
    (void)in_sizes; (void)n_in; (void)out_size; (void)ws_size;
    const int*   value   = (const int*)d_in[0];
    const int*   depth   = (const int*)d_in[1];
    /* d_in[2] = pos (unused by reference) */
    const float* emb1_w  = (const float*)d_in[3];
    const float* emb2_w  = (const float*)d_in[4];
    const float* conv1_w = (const float*)d_in[5];
    const float* conv1_b = (const float*)d_in[6];
    const float* conv2_w = (const float*)d_in[7];
    const float* conv2_b = (const float*)d_in[8];
    float*       out     = (float*)d_out;

    char* ws = (char*)d_ws;
    size_t off = 0;
    auto take = [&](size_t bytes) -> char* {
        char* p = ws + off;
        off = (off + bytes + 255) & ~(size_t)255;
        return p;
    };
    int*            maxd   = (int*)take(sizeof(int));
    int*            len1   = (int*)take(sizeof(int) * kB);
    int*            len2   = (int*)take(sizeof(int) * kB);
    int*            grank  = (int*)take((size_t)kB * kG2 * sizeof(int));
    int*            srcidx = (int*)take((size_t)kB * kL1 * sizeof(int));
    unsigned char*  tok1   = (unsigned char*)take((size_t)kB * kL1);
    unsigned char*  tok2   = (unsigned char*)take((size_t)kB * kL2);
    unsigned short* W1bf   = (unsigned short*)take((size_t)kED * kK * 2);
    unsigned short* W2bf   = (unsigned short*)take((size_t)kCD * kK * 2);
    unsigned short* E1bf   = (unsigned short*)take(128 * 2);
    unsigned short* E2bf   = (unsigned short*)take(128 * 2);
    float*          ybuf   = (float*)take((size_t)kB * kG2 * kCD * sizeof(float));

    k_init<<<1, 32, 0, stream>>>(maxd, len1, len2);
    k_maxdepth<<<512, 256, 0, stream>>>(depth, kB * kSEQ, maxd);
    k_lens<<<512, 256, 0, stream>>>(depth, maxd, len1, len2);
    k_scan<<<kB, 1024, 0, stream>>>(value, len1, len2, grank, srcidx, tok1);
    k_tok2<<<1024, 256, 0, stream>>>(value, len1, len2, tok2);
    k_wconv<<<128, 256, 0, stream>>>(conv1_w, conv2_w, emb1_w, emb2_w, W1bf, W2bf, E1bf, E2bf);
    k_gemm2<<<dim3(kG2 / 64, kB), 256, 0, stream>>>(tok2, W2bf, E2bf, conv2_b, ybuf);
    k_gemm1<<<dim3(kM1 / 64, kB), 256, 0, stream>>>(tok1, srcidx, W1bf, E1bf, ybuf, conv1_b, out);
}
